// AttentionNetwork_54460185313444
// MI455X (gfx1250) — matrix operand order repaired and checked by reading, not compiled, not run on hardware
//
#include <hip/hip_runtime.h>
#include <stdint.h>

typedef __attribute__((ext_vector_type(16))) __bf16 bf16x16;
typedef __attribute__((ext_vector_type(8)))  float  f32x8;

#define D 256
#define DH 128
#define M_TILE 128
#define N_TILE 128
#define K_PITCH 264   /* bf16 elems per sK row (256+8 pad, 16B-aligned rows) */
#define VT_PITCH 136  /* bf16 elems per sVt row (128+8) */
#define P_PITCH 136   /* bf16 elems per sP row */

#define SK_BYTES  (128 * K_PITCH * 2)    /* 67584 per buffer */
#define SVT_BYTES (256 * VT_PITCH * 2)   /* 69632 */
#define SP_BYTES  (8 * 16 * P_PITCH * 2) /* 34816 */
#define SMEM_TOTAL (2 * SK_BYTES + SVT_BYTES + SP_BYTES) /* 239616 */

__device__ __forceinline__ unsigned short f2bf(float f) {
  unsigned int u = __float_as_uint(f);
  unsigned int r = (u + 0x7FFFu + ((u >> 16) & 1u)) >> 16;  // RNE
  return (unsigned short)r;
}
__device__ __forceinline__ float bf2f(unsigned short h) {
  return __uint_as_float(((unsigned int)h) << 16);
}

union ABfrag { bf16x16 v; uint4 u[2]; };

// Wait for all of this wave's outstanding async LDS loads.
__device__ __forceinline__ void wait_async0() {
#if __has_builtin(__builtin_amdgcn_s_wait_asynccnt)
  __builtin_amdgcn_s_wait_asynccnt(0);
#else
  asm volatile("s_wait_asynccnt 0x0" ::: "memory");
#endif
}

// Issue async global->LDS fill of one 128x256 bf16 K tile (16 x b128 per lane).
__device__ __forceinline__ void issue_tile_async(const unsigned short* __restrict__ Kt,
                                                 unsigned short* sKbuf, int tid) {
#pragma unroll
  for (int it = 0; it < 16; ++it) {
    int task = it * 256 + tid;
    int row = task >> 5;   // 0..127
    int ch  = task & 31;   // 8-elem chunk
    unsigned lds = (unsigned)(uintptr_t)(sKbuf + row * K_PITCH + ch * 8);
    const void* g = (const void*)(Kt + (size_t)row * D + ch * 8);
    asm volatile("global_load_async_to_lds_b128 %0, %1, off"
                 :: "v"(lds), "v"(g) : "memory");
  }
}

// LDS->LDS transpose pass: sVt[col][row] = sK[row][col]
__device__ __forceinline__ void transpose_tile(const unsigned short* sKc,
                                               unsigned short* sVt, int tid) {
#pragma unroll
  for (int it = 0; it < 16; ++it) {
    int task = it * 256 + tid;
    int row = task >> 5;
    int ch  = task & 31;
    uint4 dv = *(const uint4*)(sKc + row * K_PITCH + ch * 8);
    unsigned short tmp[8];
    *(uint4*)tmp = dv;
#pragma unroll
    for (int j = 0; j < 8; ++j)
      sVt[(ch * 8 + j) * VT_PITCH + row] = tmp[j];
  }
}

// ---------------------------------------------------------------------------
// 1) Convert the 6 full tables to bf16 (Kbf slots: side0={x2,xn2,oh2},
//    side1={x1,xn1,oh1}) and write pass-through f32 copies to d_out.
// ---------------------------------------------------------------------------
__global__ void prep_full_kernel(const float* __restrict__ x1,  const float* __restrict__ xn1,
                                 const float* __restrict__ oh1, const float* __restrict__ x2,
                                 const float* __restrict__ xn2, const float* __restrict__ oh2,
                                 float* __restrict__ out, unsigned short* __restrict__ Kbf, int N)
{
  size_t idx = (size_t)blockIdx.x * blockDim.x + threadIdx.x;
  size_t per = (size_t)N * D;
  size_t total = 6 * per;
  if (idx >= total) return;
  int a = (int)(idx / per);
  size_t pos = idx - (size_t)a * per;
  const float* srcs[6] = {x1, xn1, oh1, x2, xn2, oh2};
  float v = srcs[a][pos];
  out[idx] = v;                          // out order: x1,xn1,oh1,x2,xn2,oh2
  int slot = (a + 3) % 6;                // Kbf order: x2,xn2,oh2,x1,xn1,oh1
  Kbf[(size_t)slot * per + pos] = f2bf(v);
}

// ---------------------------------------------------------------------------
// 2) Gather + convert Q rows: slot sc=side*3+c gathers src[side*3+c] at
//    data_batch[:,side].
// ---------------------------------------------------------------------------
__global__ void prep_q_kernel(const float* __restrict__ x1,  const float* __restrict__ xn1,
                              const float* __restrict__ oh1, const float* __restrict__ x2,
                              const float* __restrict__ xn2, const float* __restrict__ oh2,
                              const int* __restrict__ db, unsigned short* __restrict__ Qbf, int B)
{
  int blk = blockIdx.x;          // 6*B
  int sc  = blk / B;
  int b   = blk - sc * B;
  int side = sc / 3;
  const float* srcs[6] = {x1, xn1, oh1, x2, xn2, oh2};
  const float* src = srcs[sc];   // side*3 + c == sc
  int idx = db[2 * b + side];
  int t = threadIdx.x;           // 0..255 == D
  Qbf[((size_t)sc * B + b) * D + t] = f2bf(src[(size_t)idx * D + t]);
}

// ---------------------------------------------------------------------------
// 3) Flash attention: Re = softmax(Q K^T) K + Q   (per sc slot)
//    256 threads = 8 waves; wave owns 16 Q rows; K streamed in 128-row tiles
//    via double-buffered async global->LDS loads.
// ---------------------------------------------------------------------------
__global__ __launch_bounds__(256, 1) void attn_kernel(
    const unsigned short* __restrict__ Qbf,   // [6][B][D] bf16
    const unsigned short* __restrict__ Kbf,   // [6][N][D] bf16
    float* __restrict__ Re,                   // [6][B][D] f32
    int B, int N)
{
  extern __shared__ char smem[];
  unsigned short* sK0 = (unsigned short*)smem;
  unsigned short* sK1 = (unsigned short*)(smem + SK_BYTES);
  unsigned short* sVt = (unsigned short*)(smem + 2 * SK_BYTES);
  unsigned short* sP  = (unsigned short*)(smem + 2 * SK_BYTES + SVT_BYTES);

  const int sc   = blockIdx.y;
  const int tile = blockIdx.x;
  const int tid  = threadIdx.x;
  const int wave = tid >> 5;
  const int lane = tid & 31;
  const int half = lane >> 4;
  const int l16  = lane & 15;

  const unsigned short* Qg = Qbf + (size_t)sc * B * D;
  const unsigned short* Kg = Kbf + (size_t)sc * N * D;
  float* Rg = Re + (size_t)sc * B * D;

  // Preload this wave's Q rows as WMMA A-fragments (ISA 16-bit A 16x32 layout:
  // lane<16: K {0..7,16..23}; lane>=16: K {8..15,24..31})
  ABfrag qa[8];
  {
    int mrow = tile * M_TILE + wave * 16 + l16;
    const unsigned short* qrow = Qg + (size_t)mrow * D;
    int sub = half * 8;
#pragma unroll
    for (int kc = 0; kc < 8; ++kc) {
      qa[kc].u[0] = *(const uint4*)(qrow + kc * 32 + sub);
      qa[kc].u[1] = *(const uint4*)(qrow + kc * 32 + 16 + sub);
    }
  }

  f32x8 o[16];
#pragma unroll
  for (int i = 0; i < 16; ++i) o[i] = {};
  float mrowv[8], lrow[8];
#pragma unroll
  for (int r = 0; r < 8; ++r) { mrowv[r] = -1e30f; lrow[r] = 0.0f; }

  unsigned short* sPw = sP + wave * 16 * P_PITCH;
  const int ntiles = N / N_TILE;

  // Prologue: start async fill of tile 0 into buffer 0.
  issue_tile_async(Kg, sK0, tid);

  for (int nt = 0; nt < ntiles; ++nt) {
    unsigned short* sKc = (nt & 1) ? sK1 : sK0;
    unsigned short* sKn = (nt & 1) ? sK0 : sK1;

    wait_async0();        // this wave's fill of sKc has landed
    __syncthreads();      // all waves' fills visible; prev compute done

    // Build transposed copy for the P*V B-operand (LDS->LDS).
    transpose_tile(sKc, sVt, tid);

    // Kick off the next tile's async fill into the other buffer; it overlaps
    // the whole compute phase below.
    if (nt + 1 < ntiles)
      issue_tile_async(Kg + (size_t)(nt + 1) * N_TILE * D, sKn, tid);

    __syncthreads();      // sVt writes visible

    // S = Q * K^T : 8 column blocks of 16, accumulate over 8 k-chunks of 32
    f32x8 st[8];
#pragma unroll
    for (int nb = 0; nb < 8; ++nb) {
      f32x8 acc = {};
#pragma unroll
      for (int kc = 0; kc < 8; ++kc) {
        ABfrag bfr;
        const unsigned short* bp = sKc + (nb * 16 + l16) * K_PITCH + kc * 32 + half * 8;
        bfr.u[0] = *(const uint4*)(bp);
        bfr.u[1] = *(const uint4*)(bp + 16);
        acc = __builtin_amdgcn_wmma_f32_16x16x32_bf16(false, qa[kc].v, false, bfr.v,
                                                      (short)0, acc, false, false);
      }
      st[nb] = acc;
    }

    // Online softmax (row r of this wave-half; 16-lane shfl_xor reductions)
    float scv[8];
#pragma unroll
    for (int r = 0; r < 8; ++r) {
      float tm = st[0][r];
#pragma unroll
      for (int nb = 1; nb < 8; ++nb) tm = fmaxf(tm, st[nb][r]);
      tm = fmaxf(tm, __shfl_xor(tm, 1));
      tm = fmaxf(tm, __shfl_xor(tm, 2));
      tm = fmaxf(tm, __shfl_xor(tm, 4));
      tm = fmaxf(tm, __shfl_xor(tm, 8));
      float mo = mrowv[r];
      float mn = fmaxf(mo, tm);
      float scl = __expf(mo - mn);
      float rs = 0.0f;
#pragma unroll
      for (int nb = 0; nb < 8; ++nb) {
        float p = __expf(st[nb][r] - mn);
        st[nb][r] = p;
        rs += p;
      }
      rs += __shfl_xor(rs, 1);
      rs += __shfl_xor(rs, 2);
      rs += __shfl_xor(rs, 4);
      rs += __shfl_xor(rs, 8);
      lrow[r]  = lrow[r] * scl + rs;
      mrowv[r] = mn;
      scv[r]   = scl;
    }
    // Rescale accumulator
#pragma unroll
    for (int nbo = 0; nbo < 16; ++nbo) {
#pragma unroll
      for (int r = 0; r < 8; ++r) o[nbo][r] *= scv[r];
    }
    // Stage P (bf16) in this wave's private LDS region (C-layout -> row-major)
#pragma unroll
    for (int nb = 0; nb < 8; ++nb) {
#pragma unroll
      for (int r = 0; r < 8; ++r) {
        int m = r + 8 * half;
        sPw[m * P_PITCH + nb * 16 + l16] = f2bf(st[nb][r]);
      }
    }
    // O += P * V (V rows = K rows, via transposed sVt for B-operand)
#pragma unroll
    for (int kc2 = 0; kc2 < 4; ++kc2) {
      ABfrag pa;
      const unsigned short* pp = sPw + l16 * P_PITCH + kc2 * 32 + half * 8;
      pa.u[0] = *(const uint4*)(pp);
      pa.u[1] = *(const uint4*)(pp + 16);
#pragma unroll
      for (int nbo = 0; nbo < 16; ++nbo) {
        ABfrag bfr;
        const unsigned short* vp = sVt + (nbo * 16 + l16) * VT_PITCH + kc2 * 32 + half * 8;
        bfr.u[0] = *(const uint4*)(vp);
        bfr.u[1] = *(const uint4*)(vp + 16);
        o[nbo] = __builtin_amdgcn_wmma_f32_16x16x32_bf16(false, pa.v, false, bfr.v,
                                                         (short)0, o[nbo], false, false);
      }
    }
  }

  // Epilogue: Re = O / l + Q  (C-layout scatter)
  {
    int baseRow = tile * M_TILE + wave * 16;
#pragma unroll
    for (int r = 0; r < 8; ++r) {
      int m = r + 8 * half;
      int row = baseRow + m;
      float inv = 1.0f / lrow[r];
#pragma unroll
      for (int nbo = 0; nbo < 16; ++nbo) {
        int col = nbo * 16 + l16;
        float q = bf2f(Qg[(size_t)row * D + col]);
        Rg[(size_t)row * D + col] = o[nbo][r] * inv + q;
      }
    }
  }
}

// ---------------------------------------------------------------------------
// 4) MLP gate: mval[blk] = sigmoid(relu(Re_row @ W1 + b1) @ W2 + b2)
// ---------------------------------------------------------------------------
__global__ __launch_bounds__(128) void mlp_kernel(
    const float* __restrict__ Re, const float* __restrict__ W1,
    const float* __restrict__ b1, const float* __restrict__ W2,
    const float* __restrict__ b2, float* __restrict__ mval, int B)
{
  __shared__ float srow[D];
  __shared__ float sred[DH];
  int blk = blockIdx.x;   // sc*B + b
  int t = threadIdx.x;    // 0..127
  const float* row = Re + (size_t)blk * D;
  srow[t]       = row[t];
  srow[t + 128] = row[t + 128];
  __syncthreads();
  float h = 0.0f;
#pragma unroll 8
  for (int d = 0; d < D; ++d) h += srow[d] * W1[d * DH + t];
  h = fmaxf(h + b1[t], 0.0f);
  sred[t] = h * W2[t];
  __syncthreads();
  for (int s = 64; s > 0; s >>= 1) {
    if (t < s) sred[t] += sred[t + s];
    __syncthreads();
  }
  if (t == 0) {
    float logit = sred[0] + b2[0];
    mval[blk] = 1.0f / (1.0f + __expf(-logit));
  }
}

// ---------------------------------------------------------------------------
// 5) Channel softmax + row scatter-scale into d_out.
// ---------------------------------------------------------------------------
__global__ void scatter_kernel(const float* __restrict__ xn1, const float* __restrict__ oh1,
                               const float* __restrict__ xn2, const float* __restrict__ oh2,
                               const int* __restrict__ db, const float* __restrict__ mval,
                               float* __restrict__ out, int B, int N)
{
  int b = blockIdx.x;
  int side = blockIdx.y;
  int t = threadIdx.x;  // 0..255
  float e0 = mval[(side * 3 + 0) * B + b];
  float e1 = mval[(side * 3 + 1) * B + b];
  float e2 = mval[(side * 3 + 2) * B + b];
  float mx = fmaxf(e0, fmaxf(e1, e2));
  float s0 = __expf(e0 - mx), s1 = __expf(e1 - mx), s2 = __expf(e2 - mx);
  float inv = 1.0f / (s0 + s1 + s2);
  float k1 = s1 * inv, k2 = s2 * inv;
  int idx = db[2 * b + side];
  size_t per = (size_t)N * D;
  size_t off = (size_t)idx * D + t;
  if (side == 0) {
    out[1 * per + off] = xn1[off] * k1;   // x_name1_out
    out[2 * per + off] = oh1[off] * k2;   // onehot1_out
  } else {
    out[4 * per + off] = xn2[off] * k1;   // x_name2_out
    out[5 * per + off] = oh2[off] * k2;   // onehot2_out
  }
}

// ---------------------------------------------------------------------------
extern "C" void kernel_launch(void* const* d_in, const int* in_sizes, int n_in,
                              void* d_out, int out_size, void* d_ws, size_t ws_size,
                              hipStream_t stream)
{
  const float* x1  = (const float*)d_in[0];
  const float* xn1 = (const float*)d_in[1];
  const float* oh1 = (const float*)d_in[2];
  const float* x2  = (const float*)d_in[3];
  const float* xn2 = (const float*)d_in[4];
  const float* oh2 = (const float*)d_in[5];
  const float* W1  = (const float*)d_in[6];
  const float* b1  = (const float*)d_in[7];
  const float* W2  = (const float*)d_in[8];
  const float* b2  = (const float*)d_in[9];
  const int*   db  = (const int*)d_in[10];

  const int N = in_sizes[0] / D;    // 16000
  const int B = in_sizes[10] / 2;   // 4096

  char* ws = (char*)d_ws;
  size_t per = (size_t)N * D;
  unsigned short* Kbf = (unsigned short*)ws;                               // 6*per bf16
  unsigned short* Qbf = (unsigned short*)(ws + 6 * per * 2);               // 6*B*D bf16
  float* Re   = (float*)(ws + 6 * per * 2 + (size_t)6 * B * D * 2);        // 6*B*D f32
  float* mval = (float*)((char*)Re + (size_t)6 * B * D * 4);               // 6*B f32
  float* out  = (float*)d_out;

  {
    size_t total = 6 * per;
    int blocks = (int)((total + 255) / 256);
    prep_full_kernel<<<blocks, 256, 0, stream>>>(x1, xn1, oh1, x2, xn2, oh2, out, Kbf, N);
  }
  prep_q_kernel<<<6 * B, 256, 0, stream>>>(x1, xn1, oh1, x2, xn2, oh2, db, Qbf, B);
  {
    dim3 grid(B / M_TILE, 6);
    attn_kernel<<<grid, 256, SMEM_TOTAL, stream>>>(Qbf, Kbf, Re, B, N);
  }
  mlp_kernel<<<6 * B, 128, 0, stream>>>(Re, W1, b1, W2, b2, mval, B);
  {
    dim3 grid(B, 2);
    scatter_kernel<<<grid, 256, 0, stream>>>(xn1, oh1, xn2, oh2, db, mval, out, B, N);
  }
}
